// VectorQuantizer_24412594110477
// MI455X (gfx1250) — matrix operand order repaired and checked by reading, not compiled, not run on hardware
//
#include <hip/hip_runtime.h>

typedef __attribute__((ext_vector_type(16))) __bf16 v16bf;
typedef __attribute__((ext_vector_type(8)))  float  v8f;

#define EMBED_DIM 256
#define N_EMBED   8192
#define N_ROWS    32768              // 32*32*32
#define M_TILE    128                // rows per workgroup (8 waves * 16)
#define NCHUNK    32                 // codes per LDS buffer
#define NCHUNKS   (N_EMBED / NCHUNK) // 256
#define LDS_STRIDE (EMBED_DIM + 8)   // bf16 elems; 528 B row -> conflict-free ds_load_b128

// ---------------- kernel 1: ||e||^2 + bf16 codebook into workspace ----------
// Run once; both products are reused by all 256 workgroups of kernel 2.
__global__ __launch_bounds__(256)
void vq_prep_kernel(const float* __restrict__ emb,
                    float* __restrict__ enorm,
                    __bf16* __restrict__ ebf) {
    int j = blockIdx.x * blockDim.x + threadIdx.x;
    if (j >= N_EMBED) return;
    const float* row = emb + (size_t)j * EMBED_DIM;
    __bf16*      dst = ebf + (size_t)j * EMBED_DIM;
    float s = 0.f;
#pragma unroll 4
    for (int k = 0; k < EMBED_DIM; k += 4) {
        float4 v = *(const float4*)(row + k);
        s += v.x * v.x + v.y * v.y + v.z * v.z + v.w * v.w;
        dst[k]     = (__bf16)v.x;
        dst[k + 1] = (__bf16)v.y;
        dst[k + 2] = (__bf16)v.z;
        dst[k + 3] = (__bf16)v.w;
    }
    enorm[j] = s;
}

// Async-copy one 32-code bf16 chunk (16 KB) into an LDS buffer.
// 8 threads/code, 64 B (4 x b128) per thread; ASYNCcnt += 4 per thread.
// Instruction offset applies to BOTH the LDS and global addresses (ISA 10.x).
__device__ __forceinline__ void stage_async(const __bf16* __restrict__ ebf,
                                            __bf16* dst_base, int c0, int tid) {
    const int cl   = tid >> 3;          // local code 0..31
    const int kseg = (tid & 7) * 32;    // 32 bf16 = 64 B segment
    uint32_t lds   = (uint32_t)(uintptr_t)(dst_base + cl * LDS_STRIDE + kseg);
    uint32_t voff  = (uint32_t)(((c0 + cl) * EMBED_DIM + kseg) * sizeof(__bf16));
    uint64_t sbase = (uint64_t)(uintptr_t)ebf;
    asm volatile(
        "global_load_async_to_lds_b128 %0, %1, %2\n\t"
        "global_load_async_to_lds_b128 %0, %1, %2 offset:16\n\t"
        "global_load_async_to_lds_b128 %0, %1, %2 offset:32\n\t"
        "global_load_async_to_lds_b128 %0, %1, %2 offset:48"
        :: "v"(lds), "v"(voff), "s"(sbase) : "memory");
}

#define ASYNC_WAIT(n) asm volatile("s_wait_asynccnt " #n ::: "memory")

// ---------------- kernel 2: WMMA distance + argmin + gather -----------------
__global__ __launch_bounds__(256)
void vq_argmin_kernel(const float* __restrict__ hs,
                      const float* __restrict__ emb,
                      const float* __restrict__ enorm,
                      const __bf16* __restrict__ ebf,
                      float* __restrict__ zq,
                      float* __restrict__ idx_out) {
    __shared__ alignas(16) __bf16 sB[2][NCHUNK * LDS_STRIDE];
    __shared__ int sIdx[M_TILE];

    const int tid   = threadIdx.x;
    const int wave  = tid >> 5;          // 0..7
    const int lane  = tid & 31;
    const int lhalf = lane >> 4;         // 0|1
    const int l16   = lane & 15;
    const int m0    = blockIdx.x * M_TILE + wave * 16;

    // ---- Preload this wave's A tile (16 rows x 256 K) into bf16 fragments.
    // 16-bit A layout per 32-K fragment (contiguous, matches B):
    //   lanes 0-15 : elems 0-15 -> K k0..k0+15
    //   lanes 16-31: elems 0-15 -> K k0+16..k0+31
    v16bf afrag[8];
    {
        const float* aptr = hs + (size_t)(m0 + l16) * EMBED_DIM;
#pragma unroll
        for (int f = 0; f < 8; ++f) {
            const int kb = f * 32 + lhalf * 16;
#pragma unroll
            for (int i = 0; i < 8; ++i) {
                afrag[f][i]     = (__bf16)aptr[kb + i];
                afrag[f][i + 8] = (__bf16)aptr[kb + 8 + i];
            }
        }
    }

    float bestd[8];
    int   besti[8];
#pragma unroll
    for (int r = 0; r < 8; ++r) { bestd[r] = 3.4e38f; besti[r] = 0; }

    // ---- Double-buffered async pipeline: chunks c and c+1 in flight.
    stage_async(ebf, &sB[0][0], 0, tid);
    stage_async(ebf, &sB[1][0], NCHUNK, tid);

    for (int c = 0; c < NCHUNKS; ++c) {
        const int c0  = c * NCHUNK;
        const int cur = c & 1;

        // Retire this chunk's 4 async ops; leave the next chunk's 4 in flight.
        if (c + 1 < NCHUNKS) { ASYNC_WAIT(0x4); } else { ASYNC_WAIT(0x0); }
        __syncthreads();

        // ---- 2 sub-tiles of 16 codes; K=256 as 8 WMMA steps of 32.
#pragma unroll
        for (int nsub = 0; nsub < NCHUNK / 16; ++nsub) {
            const int ncol = nsub * 16 + l16;     // this lane's code column
            // 16-bit B layout per 32-K fragment (contiguous, matches A):
            //   lanes 0-15: K k0..k0+15 of col n=lane; lanes 16-31: K k0+16..k0+31
            const __bf16* bbase = &sB[cur][ncol * LDS_STRIDE + lhalf * 16];
            v16bf bfr[8];
#pragma unroll
            for (int f = 0; f < 8; ++f)           // batch the ds_loads...
                bfr[f] = *(const v16bf*)(bbase + f * 32);
            v8f acc = {};
#pragma unroll
            for (int f = 0; f < 8; ++f)           // ...then drain the WMMA chain
                acc = __builtin_amdgcn_wmma_f32_16x16x32_bf16(
                        false, afrag[f], false, bfr[f],
                        (short)0, acc, false, false);

            const int   nglob = c0 + nsub * 16 + l16;
            const float en    = enorm[nglob];
            // dist = ||e||^2 - 2 z.e   (||z||^2 constant per row -> dropped)
#pragma unroll
            for (int r = 0; r < 8; ++r) {
                float d = en - 2.0f * acc[r];
                if (d < bestd[r]) { bestd[r] = d; besti[r] = nglob; }
            }
        }
        __syncthreads();                          // everyone done reading buf[cur]
        if (c + 2 < NCHUNKS)
            stage_async(ebf, &sB[cur][0], c0 + 2 * NCHUNK, tid);
    }

    // ---- Reduce across the 16 lanes of each wave32 half (lane = n mod 16).
#pragma unroll
    for (int r = 0; r < 8; ++r) {
        float d  = bestd[r];
        int   bi = besti[r];
#pragma unroll
        for (int off = 8; off >= 1; off >>= 1) {
            float od = __shfl_xor(d, off, 16);
            int   oi = __shfl_xor(bi, off, 16);
            if (od < d || (od == d && oi < bi)) { d = od; bi = oi; }
        }
        if (l16 == 0)
            sIdx[wave * 16 + lhalf * 8 + r] = bi;   // row m0 + lhalf*8 + r
    }
    __syncthreads();

    // ---- Indices output (stored as float in the concatenated output buffer).
    if (tid < M_TILE)
        idx_out[blockIdx.x * M_TILE + tid] = (float)sIdx[tid];

    // ---- z_q gather (fp32 codebook): 128 rows x 256 floats, 2 threads/row.
    {
        const int rlocal = tid >> 1;
        const int kb     = (tid & 1) * 128;
        const int code   = sIdx[rlocal];
        const float* src = emb + (size_t)code * EMBED_DIM + kb;
        float* dst = zq + (size_t)(blockIdx.x * M_TILE + rlocal) * EMBED_DIM + kb;
#pragma unroll
        for (int k = 0; k < 128; k += 4)
            *(float4*)(dst + k) = *(const float4*)(src + k);
    }
}

extern "C" void kernel_launch(void* const* d_in, const int* in_sizes, int n_in,
                              void* d_out, int out_size, void* d_ws, size_t ws_size,
                              hipStream_t stream) {
    const float* hs  = (const float*)d_in[0];   // (32,32,32,256) fp32
    const float* emb = (const float*)d_in[1];   // (8192,256) fp32
    float*  out   = (float*)d_out;              // z_q (8388608 f32) ++ indices (32768)
    float*  enorm = (float*)d_ws;               // 8192 f32
    __bf16* ebf   = (__bf16*)(enorm + N_EMBED); // 8192x256 bf16 (4 MB)

    vq_prep_kernel<<<N_EMBED / 256, 256, 0, stream>>>(emb, enorm, ebf);
    vq_argmin_kernel<<<N_ROWS / M_TILE, 256, 0, stream>>>(
        hs, emb, enorm, ebf, out, out + (size_t)N_ROWS * EMBED_DIM);
}
